// GatedRecurrentSBP_35089882808839
// MI455X (gfx1250) — matrix operand order repaired and checked by reading, not compiled, not run on hardware
//
#include <hip/hip_runtime.h>
#include <hip/hip_bf16.h>
#include <stdint.h>
#include <math.h>
#include <initializer_list>

// ---------------- problem constants (match reference) ----------------
#define B_      128
#define DHW     49152        // D*H*W = 3*128*128
#define HD      512          // HDIM
#define ZD      64           // ZDIM
#define KSTEPS  10
#define HW      16384        // H*W
#define SPLITK  32           // encoder split-K chunks

typedef __attribute__((ext_vector_type(16))) __bf16 v16bf;
typedef __attribute__((ext_vector_type(8)))  float  v8f;

struct __attribute__((aligned(16))) U4 { unsigned int a, b, c, d; };
union Frag { v16bf v; U4 q[2]; };

__device__ __forceinline__ unsigned short f32_to_bf16(float f) {
  union { float f; unsigned int u; } cv; cv.f = f;
  unsigned int u = cv.u;
  if ((u & 0x7fffffffu) > 0x7f800000u) return (unsigned short)((u >> 16) | 0x40);
  unsigned int r = u + 0x7fffu + ((u >> 16) & 1u);   // round-to-nearest-even
  return (unsigned short)(r >> 16);
}

__device__ __forceinline__ float logsig(float x) {
  return fminf(x, 0.f) - log1pf(expf(-fabsf(x)));
}
__device__ __forceinline__ float sigm(float x) { return 1.f / (1.f + expf(-x)); }

// ---------------- conversion kernels ----------------
__global__ void k_cvt_bf16(const float* __restrict__ in, unsigned short* __restrict__ out, int n) {
  int i = blockIdx.x * blockDim.x + threadIdx.x;
  if (i < n) out[i] = f32_to_bf16(in[i]);
}

// transpose-convert: in (R x C) f32 row-major -> out (C x R) bf16 row-major
__global__ void k_tcvt_bf16(const float* __restrict__ in, unsigned short* __restrict__ out,
                            int R, int C) {
  __shared__ float tile[32][33];
  int c0 = blockIdx.x * 32, r0 = blockIdx.y * 32;
  for (int i = threadIdx.y; i < 32; i += 8) {
    int r = r0 + i, c = c0 + threadIdx.x;
    tile[i][threadIdx.x] = (r < R && c < C) ? in[(size_t)r * C + c] : 0.f;
  }
  __syncthreads();
  for (int i = threadIdx.y; i < 32; i += 8) {
    int c = c0 + i, r = r0 + threadIdx.x;
    if (c < C && r < R) out[(size_t)c * R + r] = f32_to_bf16(tile[threadIdx.x][i]);
  }
}

__global__ void k_zero_u32(unsigned int* p, int n) {
  int i = blockIdx.x * blockDim.x + threadIdx.x;
  if (i < n) p[i] = 0u;
}

// ---------------- multi-pair bf16 WMMA GEMM, software pipelined ----------------
// C (128 x N f32 row-major) op= bias + sum_i A_i (128 x K_i bf16) @ B_i
// Bt_i is N x K_i bf16 row-major (Bt[n][k] = B[k][n]) so B-fragments are
// K-contiguous per column. One wave owns a 16x64 tile (4 WMMA n-tiles).
// mode: 0 = C += sum ; 1 = split-K partial write (C + blockIdx.z*MN, Kc chunk);
//       2 = C = bias + sum
struct GemmPair { const unsigned short* A; const unsigned short* Bt; int lda, ldb, K; };
struct GemmArgs {
  GemmPair pr[3];
  int npairs;
  const float* bias;
  float* C;
  int N, mode, Kc, MN;
};

__global__ void __launch_bounds__(256)
k_wmma_gemm(GemmArgs ga) {
  const int tid  = threadIdx.x;
  const int lane = tid & 31;
  const int w    = blockIdx.x * 8 + (tid >> 5);
  const int nG   = ga.N >> 6;
  const int mT   = w / nG;                 // 0..7  (M = 128)
  const int nGrp = w % nG;
  const int half = lane >> 4, l16 = lane & 15;
  const int nbase = nGrp * 64 + l16;

  v8f acc[4];
  if (ga.mode == 2) {
#pragma unroll
    for (int t = 0; t < 4; t++) {
      float bv = ga.bias[nbase + t * 16];
#pragma unroll
      for (int e = 0; e < 8; e++) acc[t][e] = bv;
    }
  } else {
#pragma unroll
    for (int t = 0; t < 4; t++) acc[t] = v8f{};
  }

  for (int pi = 0; pi < ga.npairs; ++pi) {
    const int lda = ga.pr[pi].lda, ldb = ga.pr[pi].ldb;
    const int kb = (ga.mode == 1) ? blockIdx.z * ga.Kc : 0;
    const int kiters = ((ga.mode == 1) ? ga.Kc : ga.pr[pi].K) >> 5;
    // A fragment: row = mT*16+l16, chunks at k+8*half and k+16+8*half
    const unsigned short* Ap = ga.pr[pi].A + (size_t)(mT * 16 + l16) * lda + half * 8;
    // B fragment: row n, chunks at k+8*half and k+16+8*half
    const unsigned short* Bp = ga.pr[pi].Bt + (size_t)nbase * ldb + half * 8;

    Frag fa, fb[4];
    { const unsigned short* p = Ap + kb;
      fa.q[0] = *(const U4*)(p); fa.q[1] = *(const U4*)(p + 16); }
#pragma unroll
    for (int t = 0; t < 4; t++) {
      const unsigned short* p = Bp + (size_t)t * 16 * ldb + kb;
      fb[t].q[0] = *(const U4*)(p); fb[t].q[1] = *(const U4*)(p + 16);
    }

    int k0 = kb;
    for (int it = 0; it < kiters; ++it) {
      const int kn = (it + 1 < kiters) ? (k0 + 32) : kb;  // last-iter prefetch: benign reload
      Frag na, nb[4];
      { const unsigned short* p = Ap + kn;
        na.q[0] = *(const U4*)(p); na.q[1] = *(const U4*)(p + 16); }
#pragma unroll
      for (int t = 0; t < 4; t++) {
        const unsigned short* p = Bp + (size_t)t * 16 * ldb + kn;
        nb[t].q[0] = *(const U4*)(p); nb[t].q[1] = *(const U4*)(p + 16);
      }
#pragma unroll
      for (int t = 0; t < 4; t++)
        acc[t] = __builtin_amdgcn_wmma_f32_16x16x32_bf16(
            false, fa.v, false, fb[t].v, (short)0, acc[t], false, false);
      fa = na;
#pragma unroll
      for (int t = 0; t < 4; t++) fb[t] = nb[t];
      k0 += 32;
    }
  }

  float* Cw = (ga.mode == 1) ? (ga.C + (size_t)blockIdx.z * ga.MN) : ga.C;
#pragma unroll
  for (int e = 0; e < 8; e++) {
    const int m = mT * 16 + half * 8 + e;     // C layout: vgpr e <-> M row
    float* crow = Cw + (size_t)m * ga.N + nbase;
    if (ga.mode == 0) {
#pragma unroll
      for (int t = 0; t < 4; t++) crow[t * 16] += acc[t][e];
    } else {
#pragma unroll
      for (int t = 0; t < 4; t++) crow[t * 16] = acc[t][e];
    }
  }
}

// ---------------- encoder split-K reduction (+bias, ->bf16) ----------------
__global__ void k_reduce_henc(const float* __restrict__ part, const float* __restrict__ b_enc,
                              unsigned short* __restrict__ out) {
  int i = blockIdx.x * blockDim.x + threadIdx.x;       // 128*512
  if (i < B_ * HD) {
    float s = b_enc[i & (HD - 1)];
    for (int z = 0; z < SPLITK; z++) s += part[(size_t)z * B_ * HD + i];
    out[i] = f32_to_bf16(s);
  }
}

// ---------------- LSTM pointwise (gate order i,f,g,o) ----------------
__global__ void k_lstm_pw(const float* __restrict__ g, float* __restrict__ c,
                          unsigned short* __restrict__ h_bf) {
  int i = blockIdx.x * blockDim.x + threadIdx.x;       // 128*512
  if (i >= B_ * HD) return;
  int b = i >> 9, j = i & (HD - 1);
  const float* gr = g + (size_t)b * (4 * HD);
  float si = sigm(gr[j]);
  float sf = sigm(gr[HD + j]);
  float gg = tanhf(gr[2 * HD + j]);
  float so = sigm(gr[3 * HD + j]);
  float c2 = sf * c[i] + si * gg;
  c[i] = c2;
  h_bf[i] = f32_to_bf16(so * tanhf(c2));
}

// ---------------- heads: z sample + KL, kl straight into d_out ----------------
__global__ void k_head(const float* __restrict__ pl, const float* __restrict__ ql,
                       const float* __restrict__ eps_t, unsigned short* __restrict__ z_bf,
                       float* __restrict__ kl_out, int t) {
  int i = blockIdx.x * blockDim.x + threadIdx.x;       // 128*64
  if (i >= B_ * ZD) return;
  int b = i >> 6, d = i & (ZD - 1);
  float p_mu = pl[b * 2 * ZD + d], p_ls = pl[b * 2 * ZD + ZD + d];
  float q_mu = ql[b * 2 * ZD + d], q_ls = ql[b * 2 * ZD + ZD + d];
  float sp = sigm(p_ls) * 0.99f + 0.01f;
  float sq = sigm(q_ls) * 0.99f + 0.01f;
  float z  = q_mu + sq * eps_t[i];
  float dm = q_mu - p_mu;
  float kl = logf(sp / sq) + (sq * sq + dm * dm) / (2.f * sp * sp) - 0.5f;
  z_bf[i] = f32_to_bf16(z);
  kl_out[(size_t)b * (ZD * KSTEPS) + (size_t)d * KSTEPS + t] = kl;   // (B,Z,1,1,K)
}

// ---------------- fused decoder GEMM + stick-breaking ----------------
// dec(k,b,p) = zs[k*128+b,:] @ W_dec[:,p] + b_dec[p].  Wave = 16 batch rows x
// 16 pixels; 10 accumulators (one per k) share lane/VGPR positions, so the
// cumulative log-sigmoid over k is pure register-elementwise work.
__global__ void __launch_bounds__(256)
k_decode_sb(const unsigned short* __restrict__ zs_bf,
            const unsigned short* __restrict__ WdecT,   // 16384 x 64 bf16
            const float* __restrict__ b_dec, float* __restrict__ out) {
  const int tid  = threadIdx.x;
  const int lane = tid & 31;
  const int b0   = (tid >> 5) * 16;                 // wave -> batch tile
  const int p0   = blockIdx.x * 16;                 // pixel tile
  const int half = lane >> 4, l16 = lane & 15;
  const int n    = p0 + l16;

  Frag fb0, fb1;
  const unsigned short* bp = WdecT + (size_t)n * ZD + half * 8;
  fb0.q[0] = *(const U4*)(bp);      fb0.q[1] = *(const U4*)(bp + 16);
  fb1.q[0] = *(const U4*)(bp + 32); fb1.q[1] = *(const U4*)(bp + 48);

  const float bias = b_dec[n];
  v8f acc[KSTEPS];
#pragma unroll
  for (int t = 0; t < KSTEPS; t++) {
    v8f a;
#pragma unroll
    for (int e = 0; e < 8; e++) a[e] = bias;
    const unsigned short* ap = zs_bf + (size_t)(t * B_ + b0 + l16) * ZD + half * 8;
    Frag fa0, fa1;
    fa0.q[0] = *(const U4*)(ap);      fa0.q[1] = *(const U4*)(ap + 16);
    fa1.q[0] = *(const U4*)(ap + 32); fa1.q[1] = *(const U4*)(ap + 48);
    a = __builtin_amdgcn_wmma_f32_16x16x32_bf16(false, fa0.v, false, fb0.v, (short)0, a, false, false);
    a = __builtin_amdgcn_wmma_f32_16x16x32_bf16(false, fa1.v, false, fb1.v, (short)0, a, false, false);
    acc[t] = a;
  }

#pragma unroll
  for (int e = 0; e < 8; e++) {
    int b = b0 + half * 8 + e;
    size_t base = (size_t)b * HW + n;
    float css = 0.f;
#pragma unroll
    for (int t = 0; t < KSTEPS - 1; t++) {
      float d = acc[t][e];
      out[(size_t)t * (B_ * HW) + base] = css + logsig(d);
      css += logsig(-d);
    }
    out[(size_t)(KSTEPS - 1) * (B_ * HW) + base] = css;
  }
}

// ---------------- host orchestration ----------------
extern "C" void kernel_launch(void* const* d_in, const int* in_sizes, int n_in,
                              void* d_out, int out_size, void* d_ws, size_t ws_size,
                              hipStream_t stream) {
  (void)in_sizes; (void)n_in; (void)out_size; (void)ws_size;
  const float* x      = (const float*)d_in[0];
  const float* eps_q  = (const float*)d_in[1];
  const float* W_enc  = (const float*)d_in[2];
  const float* b_enc  = (const float*)d_in[3];
  const float* Wq_ih  = (const float*)d_in[4];
  const float* Wq_hh  = (const float*)d_in[5];
  const float* bq     = (const float*)d_in[6];
  const float* Wq_lin = (const float*)d_in[7];
  const float* bq_lin = (const float*)d_in[8];
  const float* Wp_ih  = (const float*)d_in[9];
  const float* Wp_hh  = (const float*)d_in[10];
  const float* bp     = (const float*)d_in[11];
  const float* Wp_lin = (const float*)d_in[12];
  const float* bp_lin = (const float*)d_in[13];
  const float* W_dec  = (const float*)d_in[14];
  const float* b_dec  = (const float*)d_in[15];
  float* out = (float*)d_out;
  float* kl_out = out + (size_t)KSTEPS * B_ * HW;

  // ---- workspace carve (256B aligned) ----
  char* ws = (char*)d_ws;
  size_t off = 0;
  auto carve = [&](size_t bytes) -> char* {
    char* p = ws + off;
    off = (off + bytes + 255) & ~(size_t)255;
    return p;
  };
  unsigned short* x_bf    = (unsigned short*)carve((size_t)B_ * DHW * 2);
  unsigned short* WencT   = (unsigned short*)carve((size_t)DHW * HD * 2);
  unsigned short* Wqih_b  = (unsigned short*)carve((size_t)4 * HD * 2 * HD * 2);
  unsigned short* Wqhh_b  = (unsigned short*)carve((size_t)4 * HD * HD * 2);
  unsigned short* Wpih_b  = (unsigned short*)carve((size_t)4 * HD * ZD * 2);
  unsigned short* Wphh_b  = (unsigned short*)carve((size_t)4 * HD * HD * 2);
  unsigned short* WqlinT  = (unsigned short*)carve((size_t)2 * ZD * HD * 2);
  unsigned short* WplinT  = (unsigned short*)carve((size_t)2 * ZD * HD * 2);
  unsigned short* WdecT   = (unsigned short*)carve((size_t)HW * ZD * 2);
  float*          hpart   = (float*)carve((size_t)SPLITK * B_ * HD * 4);
  unsigned short* henc_bf = (unsigned short*)carve((size_t)B_ * HD * 2);
  unsigned short* qh_bf   = (unsigned short*)carve((size_t)B_ * HD * 2);
  unsigned short* ph_bf   = (unsigned short*)carve((size_t)B_ * HD * 2);
  float*          qc      = (float*)carve((size_t)B_ * HD * 4);
  float*          pc      = (float*)carve((size_t)B_ * HD * 4);
  float*          gq      = (float*)carve((size_t)B_ * 4 * HD * 4);
  float*          gp      = (float*)carve((size_t)B_ * 4 * HD * 4);
  float*          pl      = (float*)carve((size_t)B_ * 2 * ZD * 4);
  float*          ql      = (float*)carve((size_t)B_ * 2 * ZD * 4);
  unsigned short* zs_bf   = (unsigned short*)carve((size_t)KSTEPS * B_ * ZD * 2);

  auto cvt = [&](const float* in, unsigned short* o, int n) {
    k_cvt_bf16<<<(n + 255) / 256, 256, 0, stream>>>(in, o, n);
  };
  // straight conversions (row-major already matches Bt layout: Bt[n][k] = W[n][k])
  cvt(x,     x_bf,   B_ * DHW);
  cvt(Wq_ih, Wqih_b, 4 * HD * 2 * HD);
  cvt(Wq_hh, Wqhh_b, 4 * HD * HD);
  cvt(Wp_ih, Wpih_b, 4 * HD * ZD);
  cvt(Wp_hh, Wphh_b, 4 * HD * HD);
  // transposed conversions (Bt[n][k] = W[k][n])
  k_tcvt_bf16<<<dim3(HD / 32, DHW / 32), dim3(32, 8), 0, stream>>>(W_enc, WencT, DHW, HD);
  k_tcvt_bf16<<<dim3(HW / 32, ZD / 32),  dim3(32, 8), 0, stream>>>(W_dec, WdecT, ZD, HW);
  k_tcvt_bf16<<<dim3(4, 16), dim3(32, 8), 0, stream>>>(Wq_lin, WqlinT, HD, 2 * ZD);
  k_tcvt_bf16<<<dim3(4, 16), dim3(32, 8), 0, stream>>>(Wp_lin, WplinT, HD, 2 * ZD);

  // zero recurrent state
  k_zero_u32<<<(B_ * HD / 2 + 255) / 256, 256, 0, stream>>>((unsigned int*)qh_bf, B_ * HD / 2);
  k_zero_u32<<<(B_ * HD / 2 + 255) / 256, 256, 0, stream>>>((unsigned int*)ph_bf, B_ * HD / 2);
  k_zero_u32<<<(B_ * HD + 255) / 256, 256, 0, stream>>>((unsigned int*)qc, B_ * HD);
  k_zero_u32<<<(B_ * HD + 255) / 256, 256, 0, stream>>>((unsigned int*)pc, B_ * HD);

  auto gemm = [&](float* C, int N, const float* bias, int mode,
                  std::initializer_list<GemmPair> prs) {
    GemmArgs ga{};
    int i = 0;
    for (const GemmPair& p : prs) ga.pr[i++] = p;
    ga.npairs = i;
    ga.bias = bias; ga.C = C; ga.N = N; ga.mode = mode;
    ga.Kc = DHW / SPLITK; ga.MN = B_ * N;
    dim3 grid(N / 64, 1, (mode == 1) ? SPLITK : 1);   // (M/16)*(N/64)/8 blocks == N/64
    k_wmma_gemm<<<grid, 256, 0, stream>>>(ga);
  };

  // ---- encoder: h_enc = x @ W_enc + b_enc (split-K partials + reduce) ----
  gemm(hpart, HD, nullptr, 1, {{x_bf, WencT, DHW, DHW, DHW}});
  k_reduce_henc<<<(B_ * HD + 255) / 256, 256, 0, stream>>>(hpart, b_enc, henc_bf);

  // ---- K recurrent steps (7 launches each) ----
  for (int t = 0; t < KSTEPS; t++) {
    unsigned short* z_bf_t = zs_bf + (size_t)t * B_ * ZD;
    // prior head on old ph:  pl = bp_lin + ph @ Wp_lin
    gemm(pl, 2 * ZD, bp_lin, 2, {{ph_bf, WplinT, HD, HD, HD}});
    // posterior gates: gq = bq + h_enc@Wq_ih[:,:512]^T + ph@Wq_ih[:,512:]^T + qh@Wq_hh^T
    gemm(gq, 4 * HD, bq, 2, {{henc_bf, Wqih_b,      HD, 2 * HD, HD},
                             {ph_bf,   Wqih_b + HD, HD, 2 * HD, HD},
                             {qh_bf,   Wqhh_b,      HD, HD,     HD}});
    k_lstm_pw<<<(B_ * HD + 255) / 256, 256, 0, stream>>>(gq, qc, qh_bf);
    // posterior head: ql = bq_lin + qh @ Wq_lin
    gemm(ql, 2 * ZD, bq_lin, 2, {{qh_bf, WqlinT, HD, HD, HD}});
    // z sample + KL
    k_head<<<(B_ * ZD + 255) / 256, 256, 0, stream>>>(pl, ql, eps_q + (size_t)t * B_ * ZD,
                                                      z_bf_t, kl_out, t);
    // prior gates: gp = bp + z@Wp_ih^T + ph@Wp_hh^T
    gemm(gp, 4 * HD, bp, 2, {{z_bf_t, Wpih_b, ZD, ZD, ZD},
                             {ph_bf,  Wphh_b, HD, HD, HD}});
    k_lstm_pw<<<(B_ * HD + 255) / 256, 256, 0, stream>>>(gp, pc, ph_bf);
  }

  // ---- fused decoder + stick-breaking, writes log_ms directly ----
  k_decode_sb<<<dim3(HW / 16), 256, 0, stream>>>(zs_bf, WdecT, b_dec, out);
}
